// Mamba_Block_84842783965661
// MI455X (gfx1250) — matrix operand order repaired and checked by reading, not compiled, not run on hardware
//
#include <hip/hip_runtime.h>
#include <hip/hip_bf16.h>

// ---------------------------------------------------------------------------
// Mamba block on gfx1250 (MI455X).
//  - Weights pre-converted + pre-transposed to bf16 [N][K] once per launch.
//  - GEMMs: V_WMMA_F32_16X16X32_BF16 (fp32 accumulate).
//      * A tile: software-staged (global fp32 -> v_cvt_pk_bf16_f32 -> LDS).
//      * B tile: Tensor Data Mover (tensor_load_to_lds, 6-arg clang-23 form)
//        2D tile copy with hardware padding to the conflict-free KPAD=40 LDS
//        layout; TDM's tensor_dim clamp provides N-edge zero-fill for free.
//  - Depthwise conv + SiLU fused; selective scan fp32 with global_prefetch_b8.
// ---------------------------------------------------------------------------

typedef __attribute__((ext_vector_type(4)))  float        v4f;
typedef __attribute__((ext_vector_type(8)))  float        v8f;
typedef __attribute__((ext_vector_type(8)))  __bf16       v8bf;
typedef __attribute__((ext_vector_type(16))) __bf16       v16bf;
typedef __attribute__((ext_vector_type(4)))  unsigned int u32x4;
typedef __attribute__((ext_vector_type(4)))  int          i32x4;
typedef __attribute__((ext_vector_type(8)))  int          i32x8;

#define BATCH_    4
#define SEQ_      2048
#define DIM_      1024
#define DINNER_   2048
#define DSTATE_   8
#define DTRANK_   64
#define NTOK_     (BATCH_ * SEQ_)   // 8192 rows for every GEMM

// GEMM tiling: 256 threads = 8 waves (2x4); block tile 64(M) x 128(N);
// K staged through LDS in steps of 32 (= one WMMA K depth).
#define BLK_M 64
#define BLK_N 128
#define BLK_K 32
#define KPAD  40   // LDS row length in bf16: 80 B rows (16B aligned, 20-dword
                   // stride => the 16 rows of a fragment land on distinct banks)

__device__ __forceinline__ float sigmoidf_(float x) { return 1.f / (1.f + __expf(-x)); }
__device__ __forceinline__ float softplusf_(float x) { return (x > 20.f) ? x : log1pf(__expf(x)); }

// ---------------------------------------------------------------------------
// One-time weight prep: W[K][N] fp32 -> Wt[N][K] bf16 (32x32 LDS tile transpose)
// ---------------------------------------------------------------------------
__global__ __launch_bounds__(256)
void transpose_bf16_k(const float* __restrict__ W, __bf16* __restrict__ Wt,
                      int K, int N)
{
    __shared__ float tile[32][33];
    const int k0 = blockIdx.x * 32, n0 = blockIdx.y * 32;
    const int tx = threadIdx.x & 31, ty = threadIdx.x >> 5;   // ty: 0..7
    #pragma unroll
    for (int r = ty; r < 32; r += 8) {
        const int k = k0 + r, n = n0 + tx;
        tile[r][tx] = (k < K && n < N) ? W[(size_t)k * N + n] : 0.f;
    }
    __syncthreads();
    #pragma unroll
    for (int r = ty; r < 32; r += 8) {
        const int n = n0 + r, k = k0 + tx;
        if (n < N && k < K) Wt[(size_t)n * K + k] = (__bf16)tile[tx][r];
    }
}

// ---------------------------------------------------------------------------
// C[M,N] = A[M,K] (fp32) @ Wt[N,K]^T (bf16). epilogue: 0=none,
// 1 = softplus(acc + bias[col]).  M%64==0, K%32==0, lda%4==0.
// ---------------------------------------------------------------------------
__global__ __launch_bounds__(256)
void gemm_wmma_bf16_k(const float* __restrict__ A, const __bf16* __restrict__ Bt,
                      float* __restrict__ C, const float* __restrict__ bias,
                      int N, int K, int lda, int ldc, int epilogue)
{
    __shared__ __bf16 As[BLK_M][KPAD];   //  5,120 B (K-major)
    __shared__ __bf16 Bs[BLK_N][KPAD];   // 10,240 B (K-major; filled by TDM)

    const int tid    = threadIdx.x;
    const int lane   = tid & 31;
    const int wm     = (tid >> 5) >> 2;       // 0..1
    const int wn     = (tid >> 5) & 3;        // 0..3
    const int r16    = lane & 15;
    const int half   = lane >> 4;
    const int blockM = blockIdx.y * BLK_M;
    const int blockN = blockIdx.x * BLK_N;

    // Wave-scalar flag: wave 0 drives the Tensor Data Mover.
    const bool tdm_wave = (__builtin_amdgcn_readfirstlane(tid >> 5) == 0);
    const unsigned int lds_b = (unsigned int)(size_t)&Bs[0][0];  // LDS byte offset

    // Static D# pieces (2-D tensor: groups 2/3 + trailing group zero-filled).
    // group1.w0: data_size=2B (code 1) | pad_enable | pad_interval=16dw (code 3)
    //            | pad_amount=4dw (code 3)
    const unsigned int g1w0 = (1u << 16) | (1u << 20) | (3u << 22) | (3u << 25);
    const unsigned long long tensor_base =
        (unsigned long long)(size_t)(Bt + (size_t)blockN * K);

    const v8f vzero = {0.f, 0.f, 0.f, 0.f, 0.f, 0.f, 0.f, 0.f};
    v8f acc[2][2];
    acc[0][0] = vzero; acc[0][1] = vzero; acc[1][0] = vzero; acc[1][1] = vzero;

    for (int k0 = 0; k0 < K; k0 += BLK_K) {
        // ---- B tile via TDM: 128 rows x 32 K bf16, row stride K ------------
        if (tdm_wave) {
            const unsigned long long ga = tensor_base + (unsigned long long)k0 * 2ull;
            u32x4 g0;
            g0[0] = 1u;                                   // count=1, user mode
            g0[1] = lds_b;                                // lds_addr
            g0[2] = (unsigned int)(ga & 0xFFFFFFFFull);   // global_addr[31:0]
            g0[3] = (unsigned int)((ga >> 32) & 0x01FFFFFFull) | 0x80000000u; // type=2
            i32x8 g1;
            g1[0] = (int)g1w0;
            g1[1] = (int)((unsigned)(K & 0xFFFF) << 16);            // tensor_dim0 lo
            g1[2] = (int)(((unsigned)K >> 16) | ((unsigned)(N & 0xFFFF) << 16));
            g1[3] = (int)(((unsigned)N >> 16) | (32u << 16));       // tile_dim0=32
            g1[4] = (int)128u;                                      // tile_dim1=128
            g1[5] = (int)(unsigned)K;                               // dim0_stride lo
            g1[6] = 0;
            g1[7] = 0;
            const i32x4 gz4 = {0, 0, 0, 0};
            const i32x8 gz8 = {0, 0, 0, 0, 0, 0, 0, 0};
            __builtin_amdgcn_tensor_load_to_lds(g0, g1, gz4, gz4, gz8, 0);
        }
        // ---- A tile: 64 rows x 32 K, 8 fp32 -> 8 bf16 per thread -----------
        {
            const int row = tid >> 2;              // 0..63
            const int c8  = (tid & 3) * 8;         // 0,8,16,24
            const float* src = A + (size_t)(blockM + row) * lda + k0 + c8;
            v4f f0 = *(const v4f*)src;
            v4f f1 = *(const v4f*)(src + 4);
            v8bf h;
            h[0] = (__bf16)f0.x; h[1] = (__bf16)f0.y; h[2] = (__bf16)f0.z; h[3] = (__bf16)f0.w;
            h[4] = (__bf16)f1.x; h[5] = (__bf16)f1.y; h[6] = (__bf16)f1.z; h[7] = (__bf16)f1.w;
            *(v8bf*)&As[row][c8] = h;              // aligned 16B store
        }
        if (tdm_wave) __builtin_amdgcn_s_wait_tensorcnt((short)0);
        __syncthreads();

        // ---- fragments + 4 WMMA (one full K=32 step per tile) --------------
        v16bf af[2], bfr[2];
        #pragma unroll
        for (int i = 0; i < 2; ++i) {
            const int r = wm * 32 + i * 16 + r16;
            // A layout: half 0 -> K {0..7,16..23}; half 1 -> K {8..15,24..31}
            v8bf lo = *(const v8bf*)&As[r][half * 8];
            v8bf hi = *(const v8bf*)&As[r][16 + half * 8];
            af[i] = __builtin_shufflevector(lo, hi, 0,1,2,3,4,5,6,7,8,9,10,11,12,13,14,15);
        }
        #pragma unroll
        for (int j = 0; j < 2; ++j) {
            const int c = wn * 32 + j * 16 + r16;
            // B layout: half 0 -> K {0..7,16..23}; half 1 -> K {8..15,24..31}
            v8bf lo = *(const v8bf*)&Bs[c][half * 8];
            v8bf hi = *(const v8bf*)&Bs[c][16 + half * 8];
            bfr[j] = __builtin_shufflevector(lo, hi, 0,1,2,3,4,5,6,7,8,9,10,11,12,13,14,15);
        }
        #pragma unroll
        for (int i = 0; i < 2; ++i)
            #pragma unroll
            for (int j = 0; j < 2; ++j)
                acc[i][j] = __builtin_amdgcn_wmma_f32_16x16x32_bf16(
                    false, af[i], false, bfr[j], (short)0, acc[i][j],
                    false, false);
        __syncthreads();
    }

    // ---- epilogue + store (C/D f32 layout: VGPR v -> M = v + 8*half) -------
    #pragma unroll
    for (int i = 0; i < 2; ++i) {
        #pragma unroll
        for (int j = 0; j < 2; ++j) {
            const int colBase = blockN + wn * 32 + j * 16 + r16;
            if (colBase < N) {
                const float bv = (epilogue == 1) ? bias[colBase] : 0.f;
                #pragma unroll
                for (int v = 0; v < 8; ++v) {
                    const int row = blockM + wm * 32 + i * 16 + v + (half << 3);
                    float val = acc[i][j][v];
                    if (epilogue == 1) val = softplusf_(val + bv);
                    C[(size_t)row * ldc + colBase] = val;
                }
            }
        }
    }
}

// Depthwise causal conv (K=4) over seq + SiLU. xr = xz[..., :2048] (stride 4096).
__global__ __launch_bounds__(256)
void conv_silu_k(const float* __restrict__ xz, const float* __restrict__ conv_w,
                 const float* __restrict__ conv_b, float* __restrict__ xc)
{
    const int idx = blockIdx.x * 256 + threadIdx.x;   // 0 .. 8192*512-1
    const int d4  = (idx & 511) * 4;                  // channel quad
    const int tok = idx >> 9;                         // b*2048 + t
    const int t   = tok & (SEQ_ - 1);

    v4f acc = *(const v4f*)&conv_b[d4];
    #pragma unroll
    for (int k = 0; k < 4; ++k) {
        const int tt = t - 3 + k;
        if (tt >= 0) {
            v4f xv = *(const v4f*)&xz[(size_t)(tok - 3 + k) * (2 * DINNER_) + d4];
            acc.x += conv_w[(d4 + 0) * 4 + k] * xv.x;
            acc.y += conv_w[(d4 + 1) * 4 + k] * xv.y;
            acc.z += conv_w[(d4 + 2) * 4 + k] * xv.z;
            acc.w += conv_w[(d4 + 3) * 4 + k] * xv.w;
        }
    }
    acc.x *= sigmoidf_(acc.x);
    acc.y *= sigmoidf_(acc.y);
    acc.z *= sigmoidf_(acc.z);
    acc.w *= sigmoidf_(acc.w);
    *(v4f*)&xc[(size_t)tok * DINNER_ + d4] = acc;
}

// Selective scan: one thread per (b, d); 8-state fp32 recurrence in registers,
// fused skip (+xc*D) and gate (*silu(z)); y written in place over xc.
__global__ __launch_bounds__(256)
void scan_k(const float* __restrict__ dt, float* __restrict__ xc,
            const float* __restrict__ dbl, const float* __restrict__ xz,
            const float* __restrict__ A_log, const float* __restrict__ Dv)
{
    const int g = blockIdx.x * 256 + threadIdx.x;     // 0..8191
    const int b = g >> 11;
    const int d = g & (DINNER_ - 1);

    float a[DSTATE_], h[DSTATE_];
    #pragma unroll
    for (int n = 0; n < DSTATE_; ++n) {
        a[n] = -__expf(A_log[d * DSTATE_ + n]);
        h[n] = 0.f;
    }
    const float Dd = Dv[d];
    const size_t tokbase = (size_t)b * SEQ_;

    for (int t = 0; t < SEQ_; ++t) {
        const size_t tok = tokbase + t;
        const float dtv = dt[tok * DINNER_ + d];
        const float xv  = xc[tok * DINNER_ + d];
        const float* bc = dbl + tok * (DTRANK_ + 2 * DSTATE_) + DTRANK_;
        if (t + 1 < SEQ_) {  // next-timestep prefetch (global_prefetch_b8)
            __builtin_prefetch(dbl + (tok + 1) * (DTRANK_ + 2 * DSTATE_) + DTRANK_, 0, 0);
            __builtin_prefetch(&dt[(tok + 1) * DINNER_ + d], 0, 0);
            __builtin_prefetch(&xc[(tok + 1) * DINNER_ + d], 0, 0);
        }
        const float dx = dtv * xv;
        float y = 0.f;
        #pragma unroll
        for (int n = 0; n < DSTATE_; ++n) {
            const float e = __expf(dtv * a[n]);
            h[n] = e * h[n] + dx * bc[n];
            y += h[n] * bc[DSTATE_ + n];
        }
        const float zv = xz[tok * (2 * DINNER_) + DINNER_ + d];
        y = (y + xv * Dd) * (zv * sigmoidf_(zv));
        xc[tok * DINNER_ + d] = y;
    }
}

// ---------------------------------------------------------------------------
// Scratch layout:
//   floats: xz @0 (33,554,432) | xc @33,554,432 (16,777,216)
//           dbl @50,331,648 (655,360) | dt @50,987,008 (16,777,216)
//   bf16 (after byte offset 271,056,896):
//           W_inT (4,194,304) | W_xT (163,840) | W_dtT (131,072) | W_outT (2,097,152)
//   total ~284 MB.
// ---------------------------------------------------------------------------
extern "C" void kernel_launch(void* const* d_in, const int* in_sizes, int n_in,
                              void* d_out, int out_size, void* d_ws, size_t ws_size,
                              hipStream_t stream)
{
    (void)in_sizes; (void)n_in; (void)out_size; (void)ws_size;

    const float* x      = (const float*)d_in[0];
    const float* W_in   = (const float*)d_in[1];
    const float* conv_w = (const float*)d_in[2];
    const float* conv_b = (const float*)d_in[3];
    const float* W_x    = (const float*)d_in[4];
    const float* W_dt   = (const float*)d_in[5];
    const float* b_dt   = (const float*)d_in[6];
    const float* A_log  = (const float*)d_in[7];
    const float* Dv     = (const float*)d_in[8];
    const float* W_out  = (const float*)d_in[9];
    float* out = (float*)d_out;

    float* ws  = (float*)d_ws;
    float* xz  = ws;
    float* xc  = ws + 33554432ull;
    float* dbl = ws + 50331648ull;
    float* dt  = ws + 50987008ull;

    __bf16* W_inT  = (__bf16*)((char*)d_ws + 271056896ull);
    __bf16* W_xT   = W_inT + 4194304ull;
    __bf16* W_dtT  = W_xT  + 163840ull;
    __bf16* W_outT = W_dtT + 131072ull;

    const dim3 blk(256);

    // 0) one-time weight bf16 conversion + transpose (W[K][N] -> Wt[N][K])
    transpose_bf16_k<<<dim3(DIM_ / 32, (2 * DINNER_) / 32), blk, 0, stream>>>(
        W_in, W_inT, DIM_, 2 * DINNER_);
    transpose_bf16_k<<<dim3(DINNER_ / 32, 3), blk, 0, stream>>>(
        W_x, W_xT, DINNER_, DTRANK_ + 2 * DSTATE_);
    transpose_bf16_k<<<dim3(DTRANK_ / 32, DINNER_ / 32), blk, 0, stream>>>(
        W_dt, W_dtT, DTRANK_, DINNER_);
    transpose_bf16_k<<<dim3(DINNER_ / 32, DIM_ / 32), blk, 0, stream>>>(
        W_out, W_outT, DINNER_, DIM_);

    // 1) xz = x @ W_in           (8192 x 1024 x 4096)
    gemm_wmma_bf16_k<<<dim3(2 * DINNER_ / BLK_N, NTOK_ / BLK_M), blk, 0, stream>>>(
        x, W_inT, xz, nullptr, 2 * DINNER_, DIM_, DIM_, 2 * DINNER_, 0);

    // 2) xc = silu(depthwise_conv(xz[..., :2048]))
    conv_silu_k<<<NTOK_ * (DINNER_ / 4) / 256, blk, 0, stream>>>(xz, conv_w, conv_b, xc);

    // 3) dbl = xc @ W_x          (8192 x 2048 x 80)
    gemm_wmma_bf16_k<<<dim3(1, NTOK_ / BLK_M), blk, 0, stream>>>(
        xc, W_xT, dbl, nullptr, DTRANK_ + 2 * DSTATE_, DINNER_,
        DINNER_, DTRANK_ + 2 * DSTATE_, 0);

    // 4) dt = softplus(dbl[:, :64] @ W_dt + b_dt)   (8192 x 64 x 2048)
    gemm_wmma_bf16_k<<<dim3(DINNER_ / BLK_N, NTOK_ / BLK_M), blk, 0, stream>>>(
        dbl, W_dtT, dt, b_dt, DINNER_, DTRANK_,
        DTRANK_ + 2 * DSTATE_, DINNER_, 1);

    // 5) selective scan + skip + gate; y overwrites xc
    scan_k<<<(BATCH_ * DINNER_) / 256, blk, 0, stream>>>(dt, xc, dbl, xz, A_log, Dv);

    // 6) out = y @ W_out         (8192 x 2048 x 1024)
    gemm_wmma_bf16_k<<<dim3(DIM_ / BLK_N, NTOK_ / BLK_M), blk, 0, stream>>>(
        xc, W_outT, out, nullptr, DIM_, DINNER_, DINNER_, DIM_, 0);
}
